// Pedal_26482768347625
// MI455X (gfx1250) — matrix operand order repaired and checked by reading, not compiled, not run on hardware
//
#include <hip/hip_runtime.h>
#include <hip/hip_bf16.h>

typedef __bf16 bf16_t;
typedef bf16_t v16bf __attribute__((ext_vector_type(16)));
typedef float  v8f   __attribute__((ext_vector_type(8)));

#define P_ 4
#define B_ 128
#define D_ 512
#define N_ 32768
#define K_ 10
#define SCALE_ 10.0f
#define SIM_TH_ 0.7f
#define NEG_INF_ (-1e30f)
// KL_WEIGHT = max(0.5*(1-1/60), 0.1) = 0.491666...
#define KLW_ 0.49166666666666664f

// ---------------------------------------------------------------------------
// init: neg_mask = 1, negsum = 0, r1/r2 = 0
__global__ void k_init(float* __restrict__ maskf, float* __restrict__ negsum,
                       float* __restrict__ r12) {
    int i = blockIdx.x * blockDim.x + threadIdx.x;
    if (i < N_) maskf[i] = 1.0f;
    if (i < P_ * B_) negsum[i] = 0.0f;
    if (i < 8) r12[i] = 0.0f;
}

// scatter zeros into neg_mask at cross_indices (B*K) and position (B)
__global__ void k_scatter(const int* __restrict__ crossidx,
                          const int* __restrict__ position,
                          float* __restrict__ maskf) {
    int i = blockIdx.x * blockDim.x + threadIdx.x;
    if (i < B_ * K_) {
        maskf[crossidx[i]] = 0.0f;
    } else if (i < B_ * K_ + B_) {
        maskf[position[i - B_ * K_]] = 0.0f;
    }
}

// ---------------------------------------------------------------------------
// per (p,b): fn2 = ||feature||^2 ; img = feature/||.|| ; txt = text^T/||.||
__global__ __launch_bounds__(256)
void k_norm(const float* __restrict__ feature, const float* __restrict__ textf,
            float* __restrict__ img, float* __restrict__ txt,
            float* __restrict__ fn2) {
    const int b = blockIdx.x, p = blockIdx.y, tid = threadIdx.x;
    __shared__ float rf[256], rt[256];
    const float* fr = feature + (size_t)(p * B_ + b) * D_;
    const float* tr = textf   + (size_t)(b * P_ + p) * D_;
    float sf = 0.f, st = 0.f;
    for (int d = tid; d < D_; d += 256) {
        float vf = fr[d]; sf += vf * vf;
        float vt = tr[d]; st += vt * vt;
    }
    rf[tid] = sf; rt[tid] = st;
    __syncthreads();
    for (int s = 128; s > 0; s >>= 1) {
        if (tid < s) { rf[tid] += rf[tid + s]; rt[tid] += rt[tid + s]; }
        __syncthreads();
    }
    float nf = fmaxf(sqrtf(rf[0]), 1e-12f);
    float nt = fmaxf(sqrtf(rt[0]), 1e-12f);
    float* ir = img + (size_t)(p * B_ + b) * D_;
    float* tw = txt + (size_t)(p * B_ + b) * D_;
    for (int d = tid; d < D_; d += 256) {
        ir[d] = fr[d] / nf;
        tw[d] = tr[d] / nt;
    }
    if (tid == 0) fn2[p * B_ + b] = rf[0];
}

// ---------------------------------------------------------------------------
// fp32 -> bf16 staging store + exact fp32 cn2 accumulation (one 16-float slice)
__device__ __forceinline__ void stage_store(bf16_t* __restrict__ dr,
                                            float* __restrict__ cn2row,
                                            float4 s0, float4 s1,
                                            float4 s2, float4 s3) {
    float sq = s0.x*s0.x + s0.y*s0.y + s0.z*s0.z + s0.w*s0.w
             + s1.x*s1.x + s1.y*s1.y + s1.z*s1.z + s1.w*s1.w
             + s2.x*s2.x + s2.y*s2.y + s2.z*s2.z + s2.w*s2.w
             + s3.x*s3.x + s3.y*s3.y + s3.z*s3.z + s3.w*s3.w;
    dr[0]=(bf16_t)s0.x; dr[1]=(bf16_t)s0.y; dr[2]=(bf16_t)s0.z; dr[3]=(bf16_t)s0.w;
    dr[4]=(bf16_t)s1.x; dr[5]=(bf16_t)s1.y; dr[6]=(bf16_t)s1.z; dr[7]=(bf16_t)s1.w;
    dr[8]=(bf16_t)s2.x; dr[9]=(bf16_t)s2.y; dr[10]=(bf16_t)s2.z; dr[11]=(bf16_t)s2.w;
    dr[12]=(bf16_t)s3.x; dr[13]=(bf16_t)s3.y; dr[14]=(bf16_t)s3.z; dr[15]=(bf16_t)s3.w;
    atomicAdd(cn2row, sq);
}

// Big GEMM + fused distance/exp/mask epilogue, double-buffered LDS staging.
// grid (N/128, P), 256 threads (8 waves). Wave w owns M-strip [16w,16w+16).
__global__ __launch_bounds__(256)
void k_neg_gemm(const float* __restrict__ feature,
                const float* __restrict__ centers,
                const float* __restrict__ fn2g,
                const float* __restrict__ maskf,
                float* __restrict__ negsum) {
    __shared__ __align__(64) bf16_t Bs[2][128][32];
    __shared__ float cn2s[128];

    const int p    = blockIdx.y;
    const int n0   = blockIdx.x * 128;
    const int tid  = threadIdx.x;
    const int lane = tid & 31;
    const int w    = tid >> 5;
    const int m_base = w * 16;
    const int hi   = lane >> 4;   // 0 or 1
    const int lq   = lane & 15;

    if (tid < 128) cn2s[tid] = 0.0f;

    v8f acc[8];
#pragma unroll
    for (int t = 0; t < 8; ++t) {
        v8f z = {0.f, 0.f, 0.f, 0.f, 0.f, 0.f, 0.f, 0.f};
        acc[t] = z;
    }

    const int srow = tid >> 1;            // 0..127
    const int scol = (tid & 1) * 16;      // 0 or 16
    const float* cbase = centers + ((size_t)p * N_ + (n0 + srow)) * D_ + scol;
    const float* ap_row = feature + (size_t)(p * B_ + m_base + lq) * D_ + (hi << 4);
    bf16_t* my_stage0 = &Bs[0][srow][scol];
    bf16_t* my_stage1 = &Bs[1][srow][scol];
    float*  my_cn2    = &cn2s[srow];

    // prologue: stage chunk 0 into buffer 0
    {
        const float4* src = (const float4*)cbase;
        float4 s0 = src[0], s1 = src[1], s2 = src[2], s3 = src[3];
        stage_store(my_stage0, my_cn2, s0, s1, s2, s3);
    }
    __syncthreads();

    const int NCHUNK = D_ / 32;           // 16
    for (int c = 0; c < NCHUNK; ++c) {
        const int k0  = c * 32;
        const int buf = c & 1;

        // issue global loads for next chunk first (latency hidden under WMMAs)
        float4 n0v, n1v, n2v, n3v;
        const bool have_next = (c + 1 < NCHUNK);
        if (have_next) {
            const float4* src = (const float4*)(cbase + (k0 + 32));
            n0v = src[0]; n1v = src[1]; n2v = src[2]; n3v = src[3];
        }

        // A fragment (contiguous k order, B read the same way):
        // lanes<16 -> K {0..15}; lanes>=16 -> K {16..31}
        const float* ap = ap_row + k0;
        float4 a0 = ((const float4*)ap)[0];
        float4 a1 = ((const float4*)ap)[1];
        float4 a2 = ((const float4*)(ap + 8))[0];
        float4 a3 = ((const float4*)(ap + 8))[1];
        v16bf af;
        af[0]=(bf16_t)a0.x; af[1]=(bf16_t)a0.y; af[2]=(bf16_t)a0.z; af[3]=(bf16_t)a0.w;
        af[4]=(bf16_t)a1.x; af[5]=(bf16_t)a1.y; af[6]=(bf16_t)a1.z; af[7]=(bf16_t)a1.w;
        af[8]=(bf16_t)a2.x; af[9]=(bf16_t)a2.y; af[10]=(bf16_t)a2.z; af[11]=(bf16_t)a2.w;
        af[12]=(bf16_t)a3.x; af[13]=(bf16_t)a3.y; af[14]=(bf16_t)a3.z; af[15]=(bf16_t)a3.w;

        const int kh = hi * 16;
#pragma unroll
        for (int nt = 0; nt < 8; ++nt) {
            v16bf bf = *(const v16bf*)(&Bs[buf][nt * 16 + lq][kh]);
            acc[nt] = __builtin_amdgcn_wmma_f32_16x16x32_bf16(
                false, af, false, bf, (short)0, acc[nt], false, false);
        }

        // convert + store next chunk into the other buffer
        if (have_next) {
            stage_store(buf ? my_stage0 : my_stage1, my_cn2, n0v, n1v, n2v, n3v);
        }
        __syncthreads();
    }

    // epilogue: d2 = fn2 + cn2 - 2*S -> exp(-10*sqrt) * mask, reduce over n
    float f2[8];
#pragma unroll
    for (int i = 0; i < 8; ++i) f2[i] = fn2g[p * B_ + m_base + hi * 8 + i];
    float esum[8] = {0.f, 0.f, 0.f, 0.f, 0.f, 0.f, 0.f, 0.f};
#pragma unroll
    for (int nt = 0; nt < 8; ++nt) {
        int ncol = nt * 16 + lq;
        float cn2v = cn2s[ncol];
        float mv = maskf[n0 + ncol];
#pragma unroll
        for (int i = 0; i < 8; ++i) {
            float d2 = fmaxf(f2[i] + cn2v - 2.0f * acc[nt][i], 1e-12f);
            esum[i] += expf(-SCALE_ * sqrtf(d2)) * mv;
        }
    }
#pragma unroll
    for (int i = 0; i < 8; ++i) {
        float v = esum[i];
        v += __shfl_xor(v, 1);
        v += __shfl_xor(v, 2);
        v += __shfl_xor(v, 4);
        v += __shfl_xor(v, 8);
        if (lq == 0) atomicAdd(&negsum[p * B_ + m_base + hi * 8 + i], v);
    }
}

// ---------------------------------------------------------------------------
// positive term: possum[p,b] = sum_k exp(-10*sqrt(max(fn2+cn2-2dot,1e-12)))
__global__ __launch_bounds__(128)
void k_pos(const float* __restrict__ feature, const float* __restrict__ centers,
           const int* __restrict__ crossidx, const float* __restrict__ fn2,
           float* __restrict__ possum) {
    const int b = blockIdx.x, p = blockIdx.y, tid = threadIdx.x;
    __shared__ float rd[128], rc[128];
    const float* f = feature + (size_t)(p * B_ + b) * D_;
    float total = 0.f;
    for (int k = 0; k < K_; ++k) {
        int idx = crossidx[b * K_ + k];
        const float* c = centers + ((size_t)p * N_ + idx) * D_;
        float dot = 0.f, c2 = 0.f;
        for (int d = tid; d < D_; d += 128) {
            float cv = c[d];
            dot += f[d] * cv;
            c2  += cv * cv;
        }
        rd[tid] = dot; rc[tid] = c2;
        __syncthreads();
        for (int s = 64; s > 0; s >>= 1) {
            if (tid < s) { rd[tid] += rd[tid + s]; rc[tid] += rc[tid + s]; }
            __syncthreads();
        }
        if (tid == 0) {
            float d2 = fmaxf(fn2[p * B_ + b] + rc[0] - 2.0f * rd[0], 1e-12f);
            total += expf(-SCALE_ * sqrtf(d2));
        }
        __syncthreads();
    }
    if (tid == 0) possum[p * B_ + b] = total;
}

// ---------------------------------------------------------------------------
// sims: img_sim[p,b,c], txt_sim[p,b,c] (already / TEMP), valid[p,b]
__global__ __launch_bounds__(128)
void k_sim(const float* __restrict__ img, const float* __restrict__ txt,
           float* __restrict__ img_sim, float* __restrict__ txt_sim,
           float* __restrict__ vsim) {
    const int b = blockIdx.x, p = blockIdx.y, tid = threadIdx.x;
    __shared__ float fi[D_], ft[D_];
    __shared__ int flag;
    const float* ir = img + (size_t)(p * B_ + b) * D_;
    const float* tr = txt + (size_t)(p * B_ + b) * D_;
    for (int d = tid; d < D_; d += 128) { fi[d] = ir[d]; ft[d] = tr[d]; }
    if (tid == 0) flag = 0;
    __syncthreads();
    const int c = tid;
    const float4* ic = (const float4*)(img + (size_t)(p * B_ + c) * D_);
    const float4* tc = (const float4*)(txt + (size_t)(p * B_ + c) * D_);
    const float4* fi4 = (const float4*)fi;
    const float4* ft4 = (const float4*)ft;
    float di = 0.f, dt = 0.f;
    for (int q = 0; q < D_ / 4; ++q) {
        float4 a = ic[q], x = fi4[q];
        di += x.x * a.x + x.y * a.y + x.z * a.z + x.w * a.w;
        float4 bq = tc[q], y = ft4[q];
        dt += y.x * bq.x + y.y * bq.y + y.z * bq.z + y.w * bq.w;
    }
    float is = di * 2.0f;   // / TEMP (0.5)
    float ts = dt * 2.0f;
    size_t o = (size_t)(p * B_ + b) * B_ + c;
    img_sim[o] = is;
    txt_sim[o] = ts;
    if (is > SIM_TH_ && ts > SIM_TH_) flag = 1;
    __syncthreads();
    if (tid == 0) vsim[p * B_ + b] = (float)flag;
}

// ---------------------------------------------------------------------------
// masked log-softmax KL per (p,b); accumulate r1[p], r2[p]
__global__ __launch_bounds__(128)
void k_kl(const float* __restrict__ img_sim, const float* __restrict__ txt_sim,
          const float* __restrict__ vsim, float* __restrict__ r12) {
    const int b = blockIdx.x, p = blockIdx.y, tid = threadIdx.x;
    __shared__ float red[128];
    const int c = tid;
    bool colm = vsim[p * B_ + c] != 0.f;
    size_t o = (size_t)(p * B_ + b) * B_ + c;
    float is = img_sim[o], ts = txt_sim[o];
    float il = colm ? is : NEG_INF_;
    float tl = colm ? ts : NEG_INF_;

    // max il
    red[tid] = il; __syncthreads();
    for (int s = 64; s > 0; s >>= 1) { if (tid < s) red[tid] = fmaxf(red[tid], red[tid + s]); __syncthreads(); }
    float mi = red[0]; __syncthreads();
    // max tl
    red[tid] = tl; __syncthreads();
    for (int s = 64; s > 0; s >>= 1) { if (tid < s) red[tid] = fmaxf(red[tid], red[tid + s]); __syncthreads(); }
    float mt = red[0]; __syncthreads();
    // sum exp(il - mi)
    red[tid] = expf(il - mi); __syncthreads();
    for (int s = 64; s > 0; s >>= 1) { if (tid < s) red[tid] += red[tid + s]; __syncthreads(); }
    float si = red[0]; __syncthreads();
    // sum exp(tl - mt)
    red[tid] = expf(tl - mt); __syncthreads();
    for (int s = 64; s > 0; s >>= 1) { if (tid < s) red[tid] += red[tid + s]; __syncthreads(); }
    float st = red[0]; __syncthreads();

    float lse_i = mi + logf(si);
    float lse_t = mt + logf(st);
    float ilp = il - lse_i, tlp = tl - lse_t;
    float k1 = colm ? expf(tlp) * (tlp - ilp) : 0.f;
    float k2 = colm ? expf(ilp) * (ilp - tlp) : 0.f;

    red[tid] = k1; __syncthreads();
    for (int s = 64; s > 0; s >>= 1) { if (tid < s) red[tid] += red[tid + s]; __syncthreads(); }
    float K1 = red[0]; __syncthreads();
    red[tid] = k2; __syncthreads();
    for (int s = 64; s > 0; s >>= 1) { if (tid < s) red[tid] += red[tid + s]; __syncthreads(); }
    float K2 = red[0];

    if (tid == 0 && vsim[p * B_ + b] != 0.f) {
        atomicAdd(&r12[p], K1);
        atomicAdd(&r12[4 + p], K2);
    }
}

// ---------------------------------------------------------------------------
__global__ void k_final(const float* __restrict__ negsum,
                        const float* __restrict__ possum,
                        const float* __restrict__ vsim,
                        const float* __restrict__ r12,
                        float* __restrict__ out) {
    if (threadIdx.x != 0 || blockIdx.x != 0) return;
    float cs = 0.f;
    for (int p = 0; p < P_; ++p) {
        float a = 0.f;
        for (int b = 0; b < B_; ++b)
            a += logf(negsum[p * B_ + b]) - logf(possum[p * B_ + b]);
        float l = a / (float)B_;
        if (l != l) l = 0.f;  // isnan -> 0
        cs += l;
    }
    cs *= (1.0f / (float)P_);
    float as = 0.f;
    for (int p = 0; p < P_; ++p) {
        int nv = 0;
        for (int b = 0; b < B_; ++b) nv += (vsim[p * B_ + b] != 0.f);
        float denom = (float)(nv > 1 ? nv : 1);
        float pp = 0.5f * (r12[p] + r12[4 + p]) / denom;
        if (nv > 0) as += pp;
    }
    out[0] = cs + KLW_ * as;
    out[1] = cs;
    out[2] = as;
}

__global__ void k_posvid(const int* __restrict__ vid,
                         const int* __restrict__ crossidx,
                         float* __restrict__ out) {
    int i = blockIdx.x * blockDim.x + threadIdx.x;
    if (i < B_ * K_) out[3 + i] = (float)vid[crossidx[i]];
}

// ---------------------------------------------------------------------------
extern "C" void kernel_launch(void* const* d_in, const int* in_sizes, int n_in,
                              void* d_out, int out_size, void* d_ws, size_t ws_size,
                              hipStream_t stream) {
    const float* feature  = (const float*)d_in[0];
    const float* textf    = (const float*)d_in[1];
    const float* centers  = (const float*)d_in[2];
    const int*   position = (const int*)d_in[3];
    const int*   crossidx = (const int*)d_in[4];
    const int*   vid      = (const int*)d_in[5];
    float* out = (float*)d_out;

    float* ws      = (float*)d_ws;
    float* img     = ws;
    float* txt     = img + (size_t)P_ * B_ * D_;
    float* fn2     = txt + (size_t)P_ * B_ * D_;
    float* negsum  = fn2 + P_ * B_;
    float* possum  = negsum + P_ * B_;
    float* r12     = possum + P_ * B_;
    float* maskf   = r12 + 8;
    float* vsim    = maskf + N_;
    float* img_sim = vsim + P_ * B_;
    float* txt_sim = img_sim + (size_t)P_ * B_ * B_;

    k_init<<<N_ / 256, 256, 0, stream>>>(maskf, negsum, r12);
    k_scatter<<<(B_ * K_ + B_ + 255) / 256, 256, 0, stream>>>(crossidx, position, maskf);
    k_norm<<<dim3(B_, P_), 256, 0, stream>>>(feature, textf, img, txt, fn2);
    k_neg_gemm<<<dim3(N_ / 128, P_), 256, 0, stream>>>(feature, centers, fn2, maskf, negsum);
    k_pos<<<dim3(B_, P_), 128, 0, stream>>>(feature, centers, crossidx, fn2, possum);
    k_sim<<<dim3(B_, P_), 128, 0, stream>>>(img, txt, img_sim, txt_sim, vsim);
    k_kl<<<dim3(B_, P_), 128, 0, stream>>>(img_sim, txt_sim, vsim, r12);
    k_final<<<1, 1, 0, stream>>>(negsum, possum, vsim, r12, out);
    k_posvid<<<(B_ * K_ + 255) / 256, 256, 0, stream>>>(vid, crossidx, out);
}
